// AttentionLayer_17454747091468
// MI455X (gfx1250) — matrix operand order repaired and checked by reading, not compiled, not run on hardware
//
#include <hip/hip_runtime.h>

// ---------------------------------------------------------------------------
// CDNA5 (gfx1250) fused causal attention layer, bf16 WMMA path.
// B=4, T=2048, C=1024, H=16, HD=64.
// ---------------------------------------------------------------------------

typedef __bf16 bf16_t;
typedef __attribute__((ext_vector_type(16))) __bf16 v16bf;
typedef __attribute__((ext_vector_type(8)))  float  v8f;
typedef __attribute__((ext_vector_type(4)))  unsigned int u32x4;
typedef __attribute__((ext_vector_type(4)))  __bf16 v4bf;

#define DEVI __device__ __forceinline__

// --- WMMA fragment gathers per CDNA5 ISA 7.12.2 (wave32) --------
// A (16x32 bf16, MxK): lane holds row M=lane%16; elements 0..7 = K(8*hi..),
// elements 8..15 = K(16+8*hi..)  -> two 16B LDS loads from the row.
DEVI v16bf load_frag_a(const bf16_t* base, int ld, int lane) {
  const int hi = lane >> 4;
  const bf16_t* row = base + (lane & 15) * ld;
  v16bf a;
  ((u32x4*)&a)[0] = *(const u32x4*)(row + 8 * hi);
  ((u32x4*)&a)[1] = *(const u32x4*)(row + 16 + 8 * hi);
  return a;
}
// B (32x16 bf16, KxN): lane holds col N=lane%16; elements 0..7 = K(8*hi..),
// elements 8..15 = K(16+8*hi..). Source tile stored [n][k] -> two 16B loads.
DEVI v16bf load_frag_b(const bf16_t* base, int ld, int lane) {
  const int hi = lane >> 4;
  const bf16_t* row = base + (lane & 15) * ld;
  v16bf b;
  ((u32x4*)&b)[0] = *(const u32x4*)(row + 8 * hi);
  ((u32x4*)&b)[1] = *(const u32x4*)(row + 16 + 8 * hi);
  return b;
}

DEVI v8f wmma_bf16(v16bf a, v16bf b, v8f c) {
  return __builtin_amdgcn_wmma_f32_16x16x32_bf16(false, a, false, b,
                                                 (short)0, c, false, false);
}

// --- fp32 -> bf16 bulk conversion ------------------------------------------
__global__ __launch_bounds__(256) void f32_to_bf16_kernel(
    const float* __restrict__ s, bf16_t* __restrict__ d, int n4) {
  int i = blockIdx.x * 256 + threadIdx.x;
  if (i < n4) {
    float4 v = ((const float4*)s)[i];
    v4bf o;
    o[0] = (bf16_t)v.x; o[1] = (bf16_t)v.y;
    o[2] = (bf16_t)v.z; o[3] = (bf16_t)v.w;
    ((v4bf*)d)[i] = o;
  }
}

// --- GEMM: out[M][N] = A[M][K] @ W[N][K]^T (+bias). 128x128 block tiles. ----
// 256 threads = 8 waves in a 4(M) x 2(N) grid; each wave: 2x4 16x16 tiles.
template <bool BIAS, bool F32OUT>
__global__ __launch_bounds__(256) void gemm_xwT(
    const bf16_t* __restrict__ A, const bf16_t* __restrict__ W,
    const float* __restrict__ bias, void* __restrict__ outv,
    int M, int N, int K) {
  constexpr int LDS_LD = 40;  // 32 + 8 pad: conflict-free b128 row gathers
  __shared__ __align__(16) bf16_t sA[128 * LDS_LD];
  __shared__ __align__(16) bf16_t sB[128 * LDS_LD];

  const int tid = threadIdx.x;
  const int lane = tid & 31, wave = tid >> 5;
  const int wm = wave >> 1, wn = wave & 1;
  const int hi = lane >> 4, nl = lane & 15;
  const int m0 = blockIdx.y * 128;
  const int n0 = blockIdx.x * 128;

  v8f acc[2][4] = {};

  for (int k0 = 0; k0 < K; k0 += 32) {
    __syncthreads();
    // 128 rows x 32 cols per tile = 512 x 16B chunks; 2 chunks per thread.
    for (int c = tid; c < 512; c += 256) {
      int r = c >> 2, cc = (c & 3) << 3;
      *(u32x4*)&sA[r * LDS_LD + cc] =
          *(const u32x4*)&A[(size_t)(m0 + r) * K + k0 + cc];
      *(u32x4*)&sB[r * LDS_LD + cc] =
          *(const u32x4*)&W[(size_t)(n0 + r) * K + k0 + cc];
    }
    __syncthreads();

    v16bf af[2], bfm[4];
#pragma unroll
    for (int i = 0; i < 2; ++i)
      af[i] = load_frag_a(&sA[(wm * 32 + i * 16) * LDS_LD], LDS_LD, lane);
#pragma unroll
    for (int j = 0; j < 4; ++j)
      bfm[j] = load_frag_b(&sB[(wn * 64 + j * 16) * LDS_LD], LDS_LD, lane);
#pragma unroll
    for (int i = 0; i < 2; ++i)
#pragma unroll
      for (int j = 0; j < 4; ++j)
        acc[i][j] = wmma_bf16(af[i], bfm[j], acc[i][j]);
  }

  // C/D layout: row = r + 8*hi, col = nl (verified in bridge doc).
#pragma unroll
  for (int i = 0; i < 2; ++i)
#pragma unroll
    for (int j = 0; j < 4; ++j)
#pragma unroll
      for (int r = 0; r < 8; ++r) {
        int row = m0 + wm * 32 + i * 16 + r + 8 * hi;
        int col = n0 + wn * 64 + j * 16 + nl;
        float v = acc[i][j][r];
        if (BIAS) v += bias[col];
        if constexpr (F32OUT)
          ((float*)outv)[(size_t)row * N + col] = v;
        else
          ((bf16_t*)outv)[(size_t)row * N + col] = (bf16_t)v;
      }
}

// --- Flash attention, causal, one (b,h,64-query-block) per workgroup. -------
// 128 threads = 4 waves; wave w owns query rows [w*16, w*16+16).
__global__ __launch_bounds__(128) void attn_kernel(
    const bf16_t* __restrict__ qkv,  // [B*T][3C], q|k|v per row
    bf16_t* __restrict__ y) {        // [B*T][C]
  constexpr int T = 2048, C = 1024, C3 = 3 * C;
  constexpr int LD = 72;  // 64 + 8 pad
  __shared__ __align__(16) bf16_t sQ[64 * LD];
  __shared__ __align__(16) bf16_t sK[64 * LD];
  __shared__ __align__(16) bf16_t sVt[64 * LD];  // transposed: [dim][key]
  __shared__ __align__(16) bf16_t sP[4][16 * LD];

  const int tid = threadIdx.x;
  const int lane = tid & 31, w = tid >> 5;
  const int hi = lane >> 4, nl = lane & 15;
  const int qi = blockIdx.x;            // query block index
  const int h = blockIdx.y, b = blockIdx.z;
  const int q0 = qi * 64;
  const size_t rowBase = (size_t)b * T;
  const int hoff = h * 64;

  // Load the 64x64 Q tile once; keep A-fragments in registers.
  for (int c = tid; c < 512; c += 128) {
    int r = c >> 3, cc = (c & 7) << 3;
    *(u32x4*)&sQ[r * LD + cc] =
        *(const u32x4*)&qkv[(rowBase + q0 + r) * C3 + hoff + cc];
  }
  __syncthreads();
  v16bf qf0 = load_frag_a(&sQ[(w * 16) * LD + 0], LD, lane);
  v16bf qf1 = load_frag_a(&sQ[(w * 16) * LD + 32], LD, lane);

  float m2[8], l[8];
  v8f o[4] = {};
#pragma unroll
  for (int r = 0; r < 8; ++r) { m2[r] = -__builtin_inff(); l[r] = 0.f; }

  const float SC = 0.125f * 1.4426950408889634f;  // 1/sqrt(64) * log2(e)

  for (int kb = 0; kb <= qi; ++kb) {
    const int k0 = kb * 64;
    __syncthreads();  // previous iteration done with sK/sVt
    for (int c = tid; c < 512; c += 128) {
      int r = c >> 3, cc = (c & 7) << 3;
      const size_t rw = (rowBase + k0 + r) * (size_t)C3 + hoff;
      *(u32x4*)&sK[r * LD + cc] = *(const u32x4*)&qkv[rw + C + cc];
      u32x4 t = *(const u32x4*)&qkv[rw + 2 * C + cc];
      const bf16_t* e = (const bf16_t*)&t;
#pragma unroll
      for (int jj = 0; jj < 8; ++jj) sVt[(cc + jj) * LD + r] = e[jj];
    }
    __syncthreads();

    // S = Q @ K^T  (K-dim = head dim 64 -> two wmma per 16x16 n-tile)
    v8f s[4];
#pragma unroll
    for (int j = 0; j < 4; ++j) {
      s[j] = v8f{};
      v16bf kf0 = load_frag_b(&sK[(j * 16) * LD + 0], LD, lane);
      v16bf kf1 = load_frag_b(&sK[(j * 16) * LD + 32], LD, lane);
      s[j] = wmma_bf16(qf0, kf0, s[j]);
      s[j] = wmma_bf16(qf1, kf1, s[j]);
    }

    // scale (base-2 domain) + causal mask on the diagonal block
    const bool diag = (kb == qi);
#pragma unroll
    for (int j = 0; j < 4; ++j)
#pragma unroll
      for (int r = 0; r < 8; ++r) {
        float v = s[j][r] * SC;
        if (diag && (k0 + j * 16 + nl) > (q0 + w * 16 + r + 8 * hi))
          v = -__builtin_inff();
        s[j][r] = v;
      }

    // online softmax; rows live in half-wave lane groups (16 lanes each)
#pragma unroll
    for (int r = 0; r < 8; ++r) {
      float v = fmaxf(fmaxf(s[0][r], s[1][r]), fmaxf(s[2][r], s[3][r]));
#pragma unroll
      for (int off = 8; off; off >>= 1) v = fmaxf(v, __shfl_xor(v, off, 32));
      float mn = fmaxf(m2[r], v);
      float al = exp2f(m2[r] - mn);
      m2[r] = mn;
      float rs = 0.f;
#pragma unroll
      for (int j = 0; j < 4; ++j) {
        float p = exp2f(s[j][r] - mn);
        rs += p;
        sP[w][(r + 8 * hi) * LD + j * 16 + nl] = (bf16_t)p;
      }
#pragma unroll
      for (int off = 8; off; off >>= 1) rs += __shfl_xor(rs, off, 32);
      l[r] = l[r] * al + rs;
#pragma unroll
      for (int j = 0; j < 4; ++j) o[j][r] *= al;
    }

    // cross-lane LDS RAW inside this wave: drain DS counter explicitly
    asm volatile("s_wait_dscnt 0x0" ::: "memory");

    // O += P @ V  (K-dim = 64 keys -> two wmma per n-tile)
    v16bf pa0 = load_frag_a(&sP[w][0], LD, lane);
    v16bf pa1 = load_frag_a(&sP[w][32], LD, lane);
#pragma unroll
    for (int j = 0; j < 4; ++j) {
      v16bf vb0 = load_frag_b(&sVt[(j * 16) * LD + 0], LD, lane);
      v16bf vb1 = load_frag_b(&sVt[(j * 16) * LD + 32], LD, lane);
      o[j] = wmma_bf16(pa0, vb0, o[j]);
      o[j] = wmma_bf16(pa1, vb1, o[j]);
    }
  }

  // normalize and emit bf16 y for the output projection
#pragma unroll
  for (int r = 0; r < 8; ++r) {
    const float inv = 1.0f / l[r];
    const size_t orow = (rowBase + q0 + w * 16 + r + 8 * hi) * (size_t)C + hoff;
#pragma unroll
    for (int j = 0; j < 4; ++j)
      y[orow + j * 16 + nl] = (bf16_t)(o[j][r] * inv);
  }
}

// ---------------------------------------------------------------------------
extern "C" void kernel_launch(void* const* d_in, const int* in_sizes, int n_in,
                              void* d_out, int out_size, void* d_ws,
                              size_t ws_size, hipStream_t stream) {
  (void)in_sizes; (void)n_in; (void)out_size; (void)ws_size;
  constexpr int Bb = 4, T = 2048, C = 1024, H = 16;
  constexpr int M = Bb * T;  // 8192

  const float* x  = (const float*)d_in[0];
  const float* wa = (const float*)d_in[1];  // [3C][C]
  const float* wp = (const float*)d_in[2];  // [C][C]
  const float* bp = (const float*)d_in[3];  // [C]

  bf16_t* ws = (bf16_t*)d_ws;
  size_t off = 0;
  bf16_t* xb   = ws + off; off += (size_t)M * C;       // 16 MB
  bf16_t* wab  = ws + off; off += (size_t)3 * C * C;   //  6 MB
  bf16_t* wpb  = ws + off; off += (size_t)C * C;       //  2 MB
  bf16_t* qkvb = ws + off; off += (size_t)M * 3 * C;   // 48 MB
  bf16_t* yb   = ws + off;                             // 16 MB

  // 1) fp32 -> bf16
  f32_to_bf16_kernel<<<(M * C / 4 + 255) / 256, 256, 0, stream>>>(x, xb, M * C / 4);
  f32_to_bf16_kernel<<<(3 * C * C / 4 + 255) / 256, 256, 0, stream>>>(wa, wab, 3 * C * C / 4);
  f32_to_bf16_kernel<<<(C * C / 4 + 255) / 256, 256, 0, stream>>>(wp, wpb, C * C / 4);

  // 2) qkv = x @ w_attn^T   [8192 x 3072]
  gemm_xwT<false, false><<<dim3(3 * C / 128, M / 128), 256, 0, stream>>>(
      xb, wab, nullptr, qkvb, M, 3 * C, C);

  // 3) flash attention -> y (bf16)
  attn_kernel<<<dim3(T / 64, H, Bb), 128, 0, stream>>>(qkvb, yb);

  // 4) out = y @ w_proj^T + b_proj   (fp32)
  gemm_xwT<true, true><<<dim3(C / 128, M / 128), 256, 0, stream>>>(
      yb, wpb, bp, d_out, M, C, C);
}
